// AtnPool_23450521436327
// MI455X (gfx1250) — matrix operand order repaired and checked by reading, not compiled, not run on hardware
//
#include <hip/hip_runtime.h>
#include <math.h>

// ---------------------------------------------------------------------------
// AtnPool for MI455X / gfx1250 (wave32, WMMA bf16 -> f32 accumulate)
//   B=32 S=2048 D=1024 H=8 DH=32 DO=128
// Kernel 1: z[b,h,s,o] = mask( gelu(feat@w1 + b1) @ w2 + b2 )   (WMMA path)
//           z stored bf16 (halves logit round-trip; fits in 192MB L2)
// Kernel 2: out[b, h*128+o] = sum_s feat[b,s,h*128+o] * softmax_s(z)[s,o]
// ---------------------------------------------------------------------------

typedef __attribute__((ext_vector_type(16))) __bf16 v16bf;
typedef __attribute__((ext_vector_type(8)))  __bf16 v8bf;
typedef __attribute__((ext_vector_type(8)))  float  v8f;

#define B_  32
#define S_  2048
#define D_  1024
#define H_  8
#define DH  32
#define DO  128

#define STILE 128          // seq rows per block (8 waves x 16)
#define KC    256          // K chunk of w1 staged in LDS
#define W1PAD (KC + 8)     // bf16 row stride for w1T (bank-conflict pad)
#define W2PAD 36           // bf16 row stride for w2T
#define GPAD  40           // bf16 row stride for gelu staging

__global__ __launch_bounds__(256)
void atnpool_gemm_kernel(const float* __restrict__ feat,
                         const int*   __restrict__ mask,
                         const float* __restrict__ w1,
                         const float* __restrict__ b1,
                         const float* __restrict__ w2,
                         const float* __restrict__ b2,
                         __bf16* __restrict__ z)
{
    __shared__ __bf16 w1t[DH * W1PAD];        // [n][k]  (transposed chunk)
    __shared__ __bf16 w2t[DO * W2PAD];        // [o][k]  (transposed)
    __shared__ __bf16 gl[8 * 16 * GPAD];      // per-wave gelu tile [16][GPAD]
    __shared__ int    msk[STILE];

    const int h    = blockIdx.x;              // h fastest -> feature tile L2 reuse
    const int s0   = blockIdx.y * STILE;
    const int b    = blockIdx.z;
    const int tid  = threadIdx.x;
    const int wave = tid >> 5;
    const int lane = tid & 31;
    const int lo16 = lane & 15;
    const int hi   = lane >> 4;               // 0: K 0-7/16-23, 1: K 8-15/24-31

    // Stage w2^T (bf16) and the mask tile once.
    for (int i = tid; i < DH * DO; i += 256) {
        int k = i >> 7, o = i & 127;          // coalesced over o
        w2t[o * W2PAD + k] = (__bf16)w2[(h * DH + k) * DO + o];
    }
    if (tid < STILE) msk[tid] = mask[b * S_ + s0 + tid];

    // A-fragment base: this lane owns row M = lo16 of its wave's 16-row tile.
    const int srow = s0 + wave * 16 + lo16;
    const float* arow_base = feat + ((size_t)b * S_ + srow) * D_ + hi * 8;

    v8f c1[2];
    #pragma unroll
    for (int i = 0; i < 2; ++i)
        #pragma unroll
        for (int j = 0; j < 8; ++j) c1[i][j] = 0.0f;

    // ---------------- GEMM1: [16,1024] x [1024,32], K chunked ----------------
    for (int kc = 0; kc < D_; kc += KC) {
        __syncthreads();                      // previous chunk fully consumed
        for (int i = tid; i < DH * KC; i += 256) {
            int kk = i >> 5, n = i & 31;      // coalesced over n
            w1t[n * W1PAD + kk] = (__bf16)w1[((size_t)h * D_ + kc + kk) * DH + n];
        }
        __syncthreads();

        if (kc + KC < D_)
            __builtin_prefetch(arow_base + kc + KC, 0, 1);

        #pragma unroll 2
        for (int kb = 0; kb < KC; kb += 32) {
            // A fragment (bf16) from global fp32: two aligned runs of 8
            const float4* a4 = (const float4*)(arow_base + kc + kb);
            float4 f0 = a4[0], f1 = a4[1], f2 = a4[4], f3 = a4[5];
            float fa[16] = { f0.x, f0.y, f0.z, f0.w, f1.x, f1.y, f1.z, f1.w,
                             f2.x, f2.y, f2.z, f2.w, f3.x, f3.y, f3.z, f3.w };
            v16bf a;
            #pragma unroll
            for (int e = 0; e < 16; ++e) a[e] = (__bf16)fa[e];

            #pragma unroll
            for (int nt = 0; nt < 2; ++nt) {
                const __bf16* bp = &w1t[(nt * 16 + lo16) * W1PAD + kb + hi * 8];
                v8bf blo = *(const v8bf*)(bp);
                v8bf bhi = *(const v8bf*)(bp + 16);
                v16bf bf;
                #pragma unroll
                for (int e = 0; e < 8; ++e) { bf[e] = blo[e]; bf[e + 8] = bhi[e]; }
                c1[nt] = __builtin_amdgcn_wmma_f32_16x16x32_bf16(
                    false, a, false, bf, (short)0, c1[nt], false, false);
            }
        }
    }

    // ---------------- bias + exact GELU, restage C-layout -> A-layout --------
    float b1v0 = b1[h * DH + lo16];
    float b1v1 = b1[h * DH + 16 + lo16];
    __bf16* g = &gl[wave * 16 * GPAD];
    #pragma unroll
    for (int nt = 0; nt < 2; ++nt) {
        float bb = nt ? b1v1 : b1v0;
        #pragma unroll
        for (int r = 0; r < 8; ++r) {
            int M = r + hi * 8;                           // C/D layout row
            float x  = c1[nt][r] + bb;
            float ge = 0.5f * x * (1.0f + erff(x * 0.70710678118654752f));
            g[M * GPAD + nt * 16 + lo16] = (__bf16)ge;
        }
    }
    __syncthreads();

    // ---------------- GEMM2: [16,32] x [32,128], single K step ---------------
    const __bf16* ga = g + lo16 * GPAD + hi * 16;
    v16bf a2;
    #pragma unroll
    for (int e = 0; e < 8; ++e) { a2[e] = ga[e]; a2[e + 8] = ga[8 + e]; }

    v8f c2[8];
    #pragma unroll
    for (int nt = 0; nt < 8; ++nt) {
        const __bf16* bp = &w2t[(nt * 16 + lo16) * W2PAD + hi * 16];
        v16bf bf;
        #pragma unroll
        for (int e = 0; e < 16; ++e) bf[e] = bp[e];
        v8f cz;
        #pragma unroll
        for (int j = 0; j < 8; ++j) cz[j] = 0.0f;
        c2[nt] = __builtin_amdgcn_wmma_f32_16x16x32_bf16(
            false, a2, false, bf, (short)0, cz, false, false);
    }

    // ---------------- bias + mask, store bf16 logits -------------------------
    __bf16* zp = z + ((size_t)(b * H_ + h) * S_ + s0 + wave * 16) * DO;
    #pragma unroll
    for (int nt = 0; nt < 8; ++nt) {
        float bb = b2[h * DO + nt * 16 + lo16];
        #pragma unroll
        for (int r = 0; r < 8; ++r) {
            int M = r + hi * 8;
            float v = c2[nt][r] + bb;
            if (msk[wave * 16 + M] == 0) v = -1e19f;
            zp[(size_t)M * DO + nt * 16 + lo16] = (__bf16)v;
        }
    }
}

// ---------------------------------------------------------------------------
// Kernel 2: softmax over S + feature-weighted reduction.
// Block = (h, b); 512 threads = 4 seq-partitions x 128 output columns.
// ---------------------------------------------------------------------------
__global__ __launch_bounds__(512)
void atnpool_softmax_kernel(const float* __restrict__ feat,
                            const __bf16* __restrict__ z,
                            float* __restrict__ out)
{
    const int h    = blockIdx.x;
    const int b    = blockIdx.y;
    const int o    = threadIdx.x & 127;
    const int part = threadIdx.x >> 7;           // 0..3
    const int scnt = S_ / 4;                     // 512
    const int s0   = part * scnt;

    const __bf16* zp = z + ((size_t)(b * H_ + h) * S_) * DO + o;
    const float*  fp = feat + (size_t)b * S_ * D_ + h * DO + o;

    __shared__ float rmax[4][128];
    __shared__ float rsum[4][128];
    __shared__ float racc[4][128];

    // Pass 1: max over this partition (4 independent streams for ILP)
    float m0 = -INFINITY, m1 = -INFINITY, m2 = -INFINITY, m3 = -INFINITY;
    for (int s = s0; s < s0 + scnt; s += 4) {
        m0 = fmaxf(m0, (float)zp[(size_t)(s + 0) * DO]);
        m1 = fmaxf(m1, (float)zp[(size_t)(s + 1) * DO]);
        m2 = fmaxf(m2, (float)zp[(size_t)(s + 2) * DO]);
        m3 = fmaxf(m3, (float)zp[(size_t)(s + 3) * DO]);
    }
    rmax[part][o] = fmaxf(fmaxf(m0, m1), fmaxf(m2, m3));
    __syncthreads();
    float m = fmaxf(fmaxf(rmax[0][o], rmax[1][o]),
                    fmaxf(rmax[2][o], rmax[3][o]));

    // Pass 2: exp-sum and feature-weighted accumulation (z is L2-resident)
    float sa0 = 0.f, sa1 = 0.f, sa2 = 0.f, sa3 = 0.f;
    float aa0 = 0.f, aa1 = 0.f, aa2 = 0.f, aa3 = 0.f;
    for (int s = s0; s < s0 + scnt; s += 4) {
        float e0 = __expf((float)zp[(size_t)(s + 0) * DO] - m);
        float e1 = __expf((float)zp[(size_t)(s + 1) * DO] - m);
        float e2 = __expf((float)zp[(size_t)(s + 2) * DO] - m);
        float e3 = __expf((float)zp[(size_t)(s + 3) * DO] - m);
        sa0 += e0; sa1 += e1; sa2 += e2; sa3 += e3;
        aa0 += fp[(size_t)(s + 0) * D_] * e0;
        aa1 += fp[(size_t)(s + 1) * D_] * e1;
        aa2 += fp[(size_t)(s + 2) * D_] * e2;
        aa3 += fp[(size_t)(s + 3) * D_] * e3;
    }
    rsum[part][o] = (sa0 + sa1) + (sa2 + sa3);
    racc[part][o] = (aa0 + aa1) + (aa2 + aa3);
    __syncthreads();

    if (part == 0) {
        float ts = (rsum[0][o] + rsum[1][o]) + (rsum[2][o] + rsum[3][o]);
        float ta = (racc[0][o] + racc[1][o]) + (racc[2][o] + racc[3][o]);
        out[(size_t)b * D_ + h * DO + o] = ta / ts;
    }
}

// ---------------------------------------------------------------------------
extern "C" void kernel_launch(void* const* d_in, const int* in_sizes, int n_in,
                              void* d_out, int out_size, void* d_ws, size_t ws_size,
                              hipStream_t stream)
{
    (void)in_sizes; (void)n_in; (void)out_size; (void)ws_size;

    const float* feat = (const float*)d_in[0];
    const int*   mask = (const int*)d_in[1];
    // d_in[2] = lengths: unused by the reference computation
    const float* w1   = (const float*)d_in[3];
    const float* b1   = (const float*)d_in[4];
    const float* w2   = (const float*)d_in[5];
    const float* b2   = (const float*)d_in[6];
    float*       out  = (float*)d_out;
    __bf16*      zbuf = (__bf16*)d_ws;  // B*H*S*DO bf16 = 128 MB logits

    dim3 g1(H_, S_ / STILE, B_);        // h fastest: feature-tile L2 reuse
    atnpool_gemm_kernel<<<g1, 256, 0, stream>>>(feat, mask, w1, b1, w2, b2, zbuf);

    dim3 g2(H_, B_);
    atnpool_softmax_kernel<<<g2, 512, 0, stream>>>(feat, zbuf, out);
}
